// PointerPegasus_16080357556617
// MI455X (gfx1250) — compile-verified, matrix operand order checked by reading, not run on hardware
//
#include <hip/hip_runtime.h>
#include <hip/hip_bf16.h>

#define T_DEC 512
#define N_SRC 1024
#define HID   1024
#define VOCAB 96103
#define HEADS 16

typedef __attribute__((ext_vector_type(16))) __bf16 v16bf;
typedef __attribute__((ext_vector_type(8)))  float  v8f;

// ---------------------------------------------------------------------------
// Kernel 1: head-average of cross attention: ca[t,n] = (1/16) * sum_h CA[h,t,n]
// ---------------------------------------------------------------------------
__global__ void head_avg_kernel(const float* __restrict__ ca_in,
                                float* __restrict__ ca_out) {
    int idx = blockIdx.x * blockDim.x + threadIdx.x;      // 0 .. T*N-1
    if (idx >= T_DEC * N_SRC) return;
    float s = 0.f;
#pragma unroll
    for (int h = 0; h < HEADS; ++h)
        s += ca_in[h * (T_DEC * N_SRC) + idx];
    ca_out[idx] = s * (1.0f / 16.0f);
}

// ---------------------------------------------------------------------------
// Kernel 2: h[512,1024] = ca[512,1024] @ enc[1024,1024]  (bf16 WMMA, f32 acc)
// Block tile 128x64, 8 waves (4x2), wave tile 32x32 = 2x2 WMMA 16x16, BK=32.
// ---------------------------------------------------------------------------
__launch_bounds__(256, 2)
__global__ void gemm_wmma_kernel(const float* __restrict__ A,   // ca
                                 const float* __restrict__ B,   // enc
                                 float* __restrict__ C) {       // h
    // Row stride 40 bf16 = 80 bytes (16B multiple -> vectorizable frag loads)
    __shared__ __bf16 As[128][40];     // A tile, row-major [m][k]
    __shared__ __bf16 Bt[64][40];      // B tile, transposed [n][k]

    const int tid   = threadIdx.x;
    const int lane  = tid & 31;
    const int wave  = tid >> 5;
    const int wm    = wave >> 1;                 // 0..3 : 32-row strip
    const int wn    = wave & 1;                  // 0..1 : 32-col strip
    const int m0    = blockIdx.y * 128;
    const int n0    = blockIdx.x * 64;
    const int khalf = lane >> 4;                 // 0 or 1
    const int lrow  = lane & 15;

    v8f acc[2][2];
#pragma unroll
    for (int i = 0; i < 2; ++i)
#pragma unroll
        for (int j = 0; j < 2; ++j)
#pragma unroll
            for (int v = 0; v < 8; ++v) acc[i][j][v] = 0.0f;

    for (int k0 = 0; k0 < HID; k0 += 32) {
        // Stage A tile: 128x32 f32 -> bf16 (4 float4 per thread)
#pragma unroll
        for (int t = 0; t < 4; ++t) {
            int i  = tid + t * 256;              // 0..1023 float4 slots
            int r  = i >> 3;                     // 8 float4 per 32-wide row
            int c4 = (i & 7) * 4;
            float4 v = *reinterpret_cast<const float4*>(&A[(m0 + r) * HID + k0 + c4]);
            As[r][c4 + 0] = (__bf16)v.x;  As[r][c4 + 1] = (__bf16)v.y;
            As[r][c4 + 2] = (__bf16)v.z;  As[r][c4 + 3] = (__bf16)v.w;
        }
        // Stage B tile transposed: 32x64 f32 -> Bt[n][k] (2 float4 per thread)
#pragma unroll
        for (int t = 0; t < 2; ++t) {
            int i  = tid + t * 256;              // 0..511 float4 slots
            int r  = i >> 4;                     // 16 float4 per 64-wide row
            int c4 = (i & 15) * 4;
            float4 v = *reinterpret_cast<const float4*>(&B[(k0 + r) * HID + n0 + c4]);
            Bt[c4 + 0][r] = (__bf16)v.x;  Bt[c4 + 1][r] = (__bf16)v.y;
            Bt[c4 + 2][r] = (__bf16)v.z;  Bt[c4 + 3][r] = (__bf16)v.w;
        }
        __syncthreads();

        // Fragment gathers per the 16-bit A layout:
        //   element e -> K = (e>>3)*16 + khalf*8 + (e&7)   (contiguous runs of 8)
        v16bf afrag[2], bfrag[2];
#pragma unroll
        for (int i = 0; i < 2; ++i) {
            const __bf16* ap = &As[wm * 32 + i * 16 + lrow][0];
#pragma unroll
            for (int e = 0; e < 16; ++e)
                afrag[i][e] = ap[(e >> 3) * 16 + khalf * 8 + (e & 7)];
        }
#pragma unroll
        for (int j = 0; j < 2; ++j) {
            const __bf16* bp = &Bt[wn * 32 + j * 16 + lrow][0];
#pragma unroll
            for (int e = 0; e < 16; ++e)
                bfrag[j][e] = bp[(e >> 3) * 16 + khalf * 8 + (e & 7)];
        }

#pragma unroll
        for (int i = 0; i < 2; ++i)
#pragma unroll
            for (int j = 0; j < 2; ++j)
                acc[i][j] = __builtin_amdgcn_wmma_f32_16x16x32_bf16(
                    /*neg_a=*/false, afrag[i], /*neg_b=*/false, bfrag[j],
                    /*c_mod=*/(short)0, acc[i][j],
                    /*reuse_a=*/false, /*reuse_b=*/false);
        __syncthreads();
    }

    // D layout: VGPR v -> M = v + 8*khalf, N = lane&15
#pragma unroll
    for (int i = 0; i < 2; ++i)
#pragma unroll
        for (int j = 0; j < 2; ++j)
#pragma unroll
            for (int v = 0; v < 8; ++v) {
                int row = m0 + wm * 32 + i * 16 + khalf * 8 + v;
                int col = n0 + wn * 32 + j * 16 + lrow;
                C[row * HID + col] = acc[i][j][v];
            }
}

// ---------------------------------------------------------------------------
// Kernel 3: gen_probs[t] = sigmoid([h | dec_h | dec_e][t,:] . w + b)
// ---------------------------------------------------------------------------
__global__ void gate_kernel(const float* __restrict__ h,
                            const float* __restrict__ dec_h,
                            const float* __restrict__ dec_e,
                            const float* __restrict__ w,
                            const float* __restrict__ b,
                            float* __restrict__ gen_ws,
                            float* __restrict__ out_gen) {
    __shared__ float red[256];
    const int t = blockIdx.x;
    const int tid = threadIdx.x;
    float s = 0.f;
    for (int j = tid; j < HID; j += 256) {
        s += h[t * HID + j]     * w[j];
        s += dec_h[t * HID + j] * w[HID + j];
        s += dec_e[t * HID + j] * w[2 * HID + j];
    }
    red[tid] = s;
    __syncthreads();
    for (int off = 128; off > 0; off >>= 1) {
        if (tid < off) red[tid] += red[tid + off];
        __syncthreads();
    }
    if (tid == 0) {
        float g = 1.0f / (1.0f + __expf(-(red[0] + b[0])));
        gen_ws[t] = g;
        out_gen[t] = g;
    }
}

// ---------------------------------------------------------------------------
// Kernel 4: one block per row t. Pass 1: online softmax stats over V=96103.
// Pass 2: out[t,v] = g * softmax(logits)[t,v].  Row stats kept for kernel 5.
// ---------------------------------------------------------------------------
__global__ void softmax_blend_kernel(const float* __restrict__ lg,
                                     const float* __restrict__ gen_ws,
                                     float* __restrict__ rowmax,
                                     float* __restrict__ rowsum,
                                     float* __restrict__ out_final) {
    __shared__ float rm[256], rs[256];
    const int t = blockIdx.x;
    const int tid = threadIdx.x;
    const float* row = lg + (size_t)t * VOCAB;

    float m = -3.402823466e38f, s = 0.f;
    for (int v = tid; v < VOCAB; v += 256) {
        float x = row[v];
        if (x > m) { s = s * __expf(m - x) + 1.0f; m = x; }
        else       { s += __expf(x - m); }
    }
    rm[tid] = m; rs[tid] = s;
    __syncthreads();
    for (int off = 128; off > 0; off >>= 1) {
        if (tid < off) {
            float m2 = rm[tid + off], s2 = rs[tid + off];
            float M  = fmaxf(rm[tid], m2);
            rs[tid]  = rs[tid] * __expf(rm[tid] - M) + s2 * __expf(m2 - M);
            rm[tid]  = M;
        }
        __syncthreads();
    }
    const float M   = rm[0];
    const float inv = 1.0f / rs[0];
    if (tid == 0) { rowmax[t] = M; rowsum[t] = rs[0]; }

    const float g = gen_ws[t];
    float* orow = out_final + (size_t)t * VOCAB;
    for (int v = tid; v < VOCAB; v += 256)
        orow[v] = g * __expf(row[v] - M) * inv;
}

// ---------------------------------------------------------------------------
// Kernel 5: copy-pointer scatter. Plain store -> duplicate ids race, matching
// the reference's unspecified .at[].set duplicate semantics.
// ---------------------------------------------------------------------------
__global__ void scatter_kernel(const float* __restrict__ lg,
                               const int* __restrict__ ids,
                               const float* __restrict__ ca,
                               const float* __restrict__ gen_ws,
                               const float* __restrict__ rowmax,
                               const float* __restrict__ rowsum,
                               float* __restrict__ out_final) {
    int idx = blockIdx.x * blockDim.x + threadIdx.x;      // 0 .. T*N-1
    if (idx >= T_DEC * N_SRC) return;
    int t = idx >> 10;
    int n = idx & (N_SRC - 1);
    int vcol = ids[n];
    float g    = gen_ws[t];
    float dist = __expf(lg[(size_t)t * VOCAB + vcol] - rowmax[t]) / rowsum[t];
    out_final[(size_t)t * VOCAB + vcol] = g * dist + (1.0f - g) * ca[idx];
}

// ---------------------------------------------------------------------------
extern "C" void kernel_launch(void* const* d_in, const int* in_sizes, int n_in,
                              void* d_out, int out_size, void* d_ws, size_t ws_size,
                              hipStream_t stream) {
    const float* cross = (const float*)d_in[0];   // [16,512,1024]
    const float* enc   = (const float*)d_in[1];   // [1024,1024]
    const float* dech  = (const float*)d_in[2];   // [512,1024]
    const float* dece  = (const float*)d_in[3];   // [512,1024]
    const float* lg    = (const float*)d_in[4];   // [512,96103]
    const int*   ids   = (const int*)d_in[5];     // [1024]
    const float* pw    = (const float*)d_in[6];   // [3072]
    const float* pb    = (const float*)d_in[7];   // [1]

    float* out = (float*)d_out;                   // [512] gen ++ [512*96103] final
    float* ws  = (float*)d_ws;
    float* ca   = ws;                             // 512*1024
    float* hmat = ca + T_DEC * N_SRC;             // 512*1024
    float* gen  = hmat + T_DEC * HID;             // 512
    float* rmax = gen + T_DEC;                    // 512
    float* rsum = rmax + T_DEC;                   // 512

    head_avg_kernel<<<(T_DEC * N_SRC) / 256, 256, 0, stream>>>(cross, ca);

    dim3 g2(HID / 64, T_DEC / 128);               // 16 x 4 blocks
    gemm_wmma_kernel<<<g2, 256, 0, stream>>>(ca, enc, hmat);

    gate_kernel<<<T_DEC, 256, 0, stream>>>(hmat, dech, dece, pw, pb, gen, out);

    softmax_blend_kernel<<<T_DEC, 256, 0, stream>>>(lg, gen, rmax, rsum, out + T_DEC);

    scatter_kernel<<<(T_DEC * N_SRC) / 256, 256, 0, stream>>>(
        lg, ids, ca, gen, rmax, rsum, out + T_DEC);
}
